// ONNX_TRT_MASK2_36240934043987
// MI455X (gfx1250) — matrix operand order repaired and checked by reading, not compiled, not run on hardware
//
#include <hip/hip_runtime.h>
#include <cstdint>

typedef __attribute__((ext_vector_type(16))) _Float16     v16h;
typedef __attribute__((ext_vector_type(8)))  float        v8f;
typedef __attribute__((ext_vector_type(4)))  unsigned int v4u;
typedef __attribute__((ext_vector_type(8)))  int          v8i;
typedef __attribute__((ext_vector_type(4)))  int          v4i;

#define Bn    4
#define Nn    25200
#define MAXD  100
#define NBn   5
#define ARn   14
#define MRn   56
#define ATT_INNER (NBn * ARn * ARn)   /* 980 */
#define HEAD  2404                     /* 4 + 1600 + 400 + 400 */

// LDS layout (bytes):
//   W16   [64][32] f16 : interp weights, row-major, padded       @ 0      (4096)
//   A16c  [5][16][32]  : gathered attn, COLUMN-major per channel @ 4096   (5120)
//   T16   [5][64][32]  : stage-1 result W*A, padded              @ 9216   (20480)
//   STAGE [980] f32    : raw attn row landed by TDM              @ 29696  (3920)
#define W_OFF     0
#define A_OFF     4096
#define T_OFF     9216
#define STAGE_OFF 29696
#define ZERO_BYTES 29696
#define SMEM_BYTES 33664

#if defined(__has_builtin)
#if __has_builtin(__builtin_amdgcn_tensor_load_to_lds) && __has_builtin(__builtin_amdgcn_s_wait_tensorcnt)
#define USE_TDM 1
#endif
#endif
#ifndef USE_TDM
#define USE_TDM 0
#endif

__global__ __launch_bounds__(256)
void mask_kernel(const float* __restrict__ attn,
                 const int*   __restrict__ det_indices,
                 const float* __restrict__ pooled_bases,
                 float*       __restrict__ out)
{
    extern __shared__ char smem[];
    _Float16* W16  = (_Float16*)(smem + W_OFF);
    _Float16* A16c = (_Float16*)(smem + A_OFF);
    _Float16* T16  = (_Float16*)(smem + T_OFF);
    float*    stg  = (float*)   (smem + STAGE_OFF);

    const int tid  = threadIdx.x;
    const int lane = tid & 31;
    const int wid  = tid >> 5;
    const int bm   = blockIdx.x;          // 0..399
    const int bIdx = bm / MAXD;
    const int det  = bm - bIdx * MAXD;

    // Warm L2 with this detection's pooled_bases slab (62720 B).
    {
        const char* pbase = (const char*)(pooled_bases + (size_t)bm * NBn * MRn * MRn);
        __builtin_prefetch(pbase + (size_t)tid * 256, 0, 0);
    }

    const int idx = det_indices[bIdx * MAXD + det];
    const float* arow = attn + ((size_t)bIdx * Nn + (size_t)idx) * ATT_INNER;

#if USE_TDM
    // One TDM DMA of the whole 980-float attn row into LDS staging,
    // issued by wave 0 while everyone else zero-fills LDS.
    if (wid == 0) {
        const uint64_t gaddr = (uint64_t)(uintptr_t)arow;
        const uint32_t laddr = (uint32_t)(size_t)(smem + STAGE_OFF); // LDS byte offset
        v4u g0;
        g0[0] = 1u;                                              // count=1 (valid user D#)
        g0[1] = laddr;                                           // lds_addr
        g0[2] = (uint32_t)(gaddr & 0xffffffffu);                 // global_addr[31:0]
        g0[3] = (uint32_t)((gaddr >> 32) & 0x01ffffffu)          // global_addr[56:32]
              | (2u << 30);                                      // type = 2 ("image")
        v8i g1;
        g1[0] = (int)(2u << 16);                 // wg_mask=0, data_size=2 (4 bytes)
        g1[1] = (int)((ATT_INNER & 0xffff) << 16);  // tensor_dim0[15:0]
        g1[2] = (int)(1u << 16);                 // tensor_dim0 hi=0 | tensor_dim1=1
        g1[3] = (int)(ATT_INNER << 16);          // tensor_dim1 hi=0 | tile_dim0=980
        g1[4] = 0;                               // tile_dim1=0 (unused), tile_dim2=0
        g1[5] = (int)ATT_INNER;                  // tensor_dim0_stride[31:0]
        g1[6] = 0;                               // stride hi / dim1_stride lo
        g1[7] = 0;
        v4i z4 = {0, 0, 0, 0};
#if __clang_major__ >= 23
        v8i z8 = {0, 0, 0, 0, 0, 0, 0, 0};
        __builtin_amdgcn_tensor_load_to_lds(g0, g1, z4, z4, z8, 0);
#else
        __builtin_amdgcn_tensor_load_to_lds(g0, g1, z4, z4, 0);
#endif
    }
#endif

    // Zero W16/A16c/T16 (contiguous first 29696 bytes).
    {
        uint32_t* z = (uint32_t*)smem;
        for (int i = tid; i < ZERO_BYTES / 4; i += 256) z[i] = 0u;
    }
    __syncthreads();

    // Bilinear weights W (56x14): half-pixel centers, scale 0.25, edge clamp.
    if (tid < MRn) {
        const int i = tid;
        float y   = (i + 0.5f) * 0.25f - 0.5f;
        float y0f = floorf(y);
        int   y0  = (int)y0f;
        float f   = y - y0f;
        int j0 = y0     < 0 ? 0 : (y0     > ARn - 1 ? ARn - 1 : y0);
        int j1 = y0 + 1 < 0 ? 0 : (y0 + 1 > ARn - 1 ? ARn - 1 : y0 + 1);
        W16[i * 32 + j0] = (_Float16)((float)W16[i * 32 + j0] + (1.0f - f));
        W16[i * 32 + j1] = (_Float16)((float)W16[i * 32 + j1] + f);
    }

#if USE_TDM
    if (wid == 0) __builtin_amdgcn_s_wait_tensorcnt(0);
#else
    for (int e = tid; e < ATT_INNER; e += 256) stg[e] = arow[e];
#endif
    __syncthreads();

    // Convert staged f32 attn row -> column-major f16 A16c[ch][col][k].
    for (int e = tid; e < ATT_INNER; e += 256) {
        int ch  = e / (ARn * ARn);
        int rem = e - ch * (ARn * ARn);
        int r   = rem / ARn;
        int c   = rem - r * ARn;
        A16c[(ch * 16 + c) * 32 + r] = (_Float16)stg[e];
    }
    __syncthreads();

    // Fragment geometry (wave32 WMMA layouts).
    const int rbase = lane & 15;             // A-operand row within tile
    const int khA   = (lane & 16) ? 8  : 0;  // A-operand K sub-base
    const int kbB   = (lane & 16) ? 16 : 0;  // unused; B loads use khA (table order)
    const int mro   = (lane & 16) ? 8  : 0;  // C/D row offset
    const int ncol  = lane & 15;             // C/D / B column

    // ---- Stage 1: T[ch](56x14) = W(56x14) x A[ch](14x14).
    // 20 WMMA units (ch,mtile) spread over all 8 waves.
    for (int u = wid; u < NBn * 4; u += 8) {
        const int mt = u & 3;
        const int ch = u >> 2;
        v16h a, b;
        const _Float16* arW = W16 + (16 * mt + rbase) * 32;
        #pragma unroll
        for (int j = 0; j < 8; ++j) { a[j] = arW[khA + j]; a[8 + j] = arW[16 + khA + j]; }
        const _Float16* bc = A16c + (ch * 16 + ncol) * 32;   // table-order K halves
        #pragma unroll
        for (int j = 0; j < 8; ++j) { b[j] = bc[khA + j]; b[8 + j] = bc[16 + khA + j]; }
        v8f c = {};
        c = __builtin_amdgcn_wmma_f32_16x16x32_f16(false, a, false, b,
                                                   (short)0, c, false, false);
        _Float16* Tm = T16 + ch * 64 * 32;
        #pragma unroll
        for (int r = 0; r < 8; ++r) {
            const int Mr = 16 * mt + mro + r;
            if (ncol < ARn) Tm[Mr * 32 + ncol] = (_Float16)c[r];
        }
    }
    __syncthreads();

    // ---- Stage 2 + fused epilogue: per output tile (mt,nt), all 5 channels
    // accumulate in registers; softmax/dot/sigmoid in registers; store global.
    const float* pb   = pooled_bases + (size_t)bm * NBn * MRn * MRn;
    float*       mout = out + HEAD + (size_t)bm * MRn * MRn;
    for (int t = wid; t < 16; t += 8) {
        const int mt = t >> 2;
        const int nt = t & 3;
        v16h b;
        // B-operand = Wt[k][n] = W[n][k]: row-major W16 rows, contiguous.
        const _Float16* br = W16 + (16 * nt + ncol) * 32;
        #pragma unroll
        for (int j = 0; j < 8; ++j) { b[j] = br[khA + j]; b[8 + j] = br[16 + khA + j]; }
        v8f d[NBn];
        #pragma unroll
        for (int ch = 0; ch < NBn; ++ch) {
            v16h a;
            const _Float16* ar = T16 + ch * 64 * 32 + (16 * mt + rbase) * 32;
            #pragma unroll
            for (int j = 0; j < 8; ++j) { a[j] = ar[khA + j]; a[8 + j] = ar[16 + khA + j]; }
            v8f z = {};
            d[ch] = __builtin_amdgcn_wmma_f32_16x16x32_f16(false, a, false, b,
                                                           (short)0, z, false, false);
        }
        #pragma unroll
        for (int r = 0; r < 8; ++r) {
            const int Mr = 16 * mt + mro + r;
            const int Nc = 16 * nt + ncol;
            if (Mr < MRn && Nc < MRn) {
                float v[NBn];
                float mx = -1e30f;
                #pragma unroll
                for (int ch = 0; ch < NBn; ++ch) { v[ch] = d[ch][r]; mx = fmaxf(mx, v[ch]); }
                float s = 0.0f;
                #pragma unroll
                for (int ch = 0; ch < NBn; ++ch) { v[ch] = __expf(v[ch] - mx); s += v[ch]; }
                const float inv = 1.0f / s;
                float acc = 0.0f;
                #pragma unroll
                for (int ch = 0; ch < NBn; ++ch)
                    acc += pb[ch * MRn * MRn + Mr * MRn + Nc] * (v[ch] * inv);
                mout[Mr * MRn + Nc] = 1.0f / (1.0f + __expf(-acc));
            }
        }
    }
}

// Pass-through head: num_det(int->f32), det_boxes, det_scores, det_classes(int->f32).
__global__ void head_kernel(const int* __restrict__ num_det,
                            const float* __restrict__ boxes,
                            const float* __restrict__ scores,
                            const int* __restrict__ classes,
                            float* __restrict__ out)
{
    const int t      = blockIdx.x * blockDim.x + threadIdx.x;
    const int stride = gridDim.x * blockDim.x;
    for (int i = t; i < Bn; i += stride)            out[i]        = (float)num_det[i];
    for (int i = t; i < Bn * MAXD * 4; i += stride) out[4 + i]    = boxes[i];
    for (int i = t; i < Bn * MAXD; i += stride)     out[1604 + i] = scores[i];
    for (int i = t; i < Bn * MAXD; i += stride)     out[2004 + i] = (float)classes[i];
}

extern "C" void kernel_launch(void* const* d_in, const int* in_sizes, int n_in,
                              void* d_out, int out_size, void* d_ws, size_t ws_size,
                              hipStream_t stream)
{
    (void)in_sizes; (void)n_in; (void)out_size; (void)d_ws; (void)ws_size;
    // 0:x0 1:attn 2:bases 3:num_det 4:det_boxes 5:det_scores 6:det_classes
    // 7:det_indices 8:pooled_bases
    const float* attn        = (const float*)d_in[1];
    const int*   num_det     = (const int*)  d_in[3];
    const float* det_boxes   = (const float*)d_in[4];
    const float* det_scores  = (const float*)d_in[5];
    const int*   det_classes = (const int*)  d_in[6];
    const int*   det_indices = (const int*)  d_in[7];
    const float* pooled      = (const float*)d_in[8];
    float*       out         = (float*)d_out;

    head_kernel<<<8, 256, 0, stream>>>(num_det, det_boxes, det_scores, det_classes, out);
    mask_kernel<<<Bn * MAXD, 256, SMEM_BYTES, stream>>>(attn, det_indices, pooled, out);
}
